// EstiFormerMultiChannelInputLayer_58600533786823
// MI455X (gfx1250) — compile-verified, matrix operand order checked by reading, not run on hardware
//
#include <hip/hip_runtime.h>
#include <hip/hip_bf16.h>

typedef __attribute__((ext_vector_type(16))) __bf16 v16bf;
typedef __attribute__((ext_vector_type(8)))  __bf16 bf16x8;
typedef __attribute__((ext_vector_type(8)))  float  v8f;
typedef __attribute__((__vector_size__(4 * sizeof(int)))) int v4i_vs;

#define CNUM 8
#define HDIM 512
#define KDIM 4096          // C*H
#define TOK  32768         // B*S
#define EPSF 1e-5f

#define MTILE 64           // tokens per workgroup
#define KCH   32           // K chunk staged in LDS (one WMMA K-step)
#define LDK   40           // padded LDS row stride (bf16 elems) = 80B, bank-conflict free

// workspace layout (bytes)
#define WS_STATS 0                          // 8 ch * 8 floats
#define WS_GW    256
#define WS_GB    (WS_GW + KDIM*4)
#define WS_BETA  (WS_GB + KDIM*4)
#define WS_TOK   (WS_BETA + KDIM*4)         // TOK*CNUM float2 (2 MB)
#define WS_BHI   (WS_TOK + TOK*CNUM*8)      // mod_w hi bf16 (4 MB)
#define WS_BLO   (WS_BHI + HDIM*KDIM*2)     // mod_w lo bf16 (4 MB)

__device__ __forceinline__ float gelu_exact(float x) {
    return 0.5f * x * (1.0f + erff(x * 0.70710678118654752440f));
}

// ---------- CDNA5 async global->LDS copy (16B per lane), with fallback ----------
#if __has_builtin(__builtin_amdgcn_global_load_async_to_lds_b128)
#define HAVE_ASYNC_LDS 1
#else
#define HAVE_ASYNC_LDS 0
#endif

__device__ __forceinline__ void cp16_async(const __bf16* g, __bf16* l) {
#if HAVE_ASYNC_LDS
    __builtin_amdgcn_global_load_async_to_lds_b128(
        (__attribute__((address_space(1))) v4i_vs*)g,
        (__attribute__((address_space(3))) v4i_vs*)l, 0, 0);
#else
    *(bf16x8*)l = *(const bf16x8*)g;
#endif
}

__device__ __forceinline__ void wait_async_copies() {
#if HAVE_ASYNC_LDS
#if __has_builtin(__builtin_amdgcn_s_wait_asynccnt)
    __builtin_amdgcn_s_wait_asynccnt(0);
#else
    asm volatile("s_wait_asynccnt 0" ::: "memory");
#endif
#endif
}

// ---------- Prep 1: per-channel stats + folded encoder constants ----------
__global__ void prep_stats_kernel(const float* __restrict__ enc_w,
                                  const float* __restrict__ enc_b,
                                  const float* __restrict__ enc_gamma,
                                  const float* __restrict__ enc_beta,
                                  float* __restrict__ stats,
                                  float* __restrict__ gw,
                                  float* __restrict__ gb,
                                  float* __restrict__ bet) {
    int c = blockIdx.x;
    int tid = threadIdx.x;
    __shared__ float red[8][5];
    __shared__ float fin[5];
    float w0 = enc_w[c*HDIM + tid],     b0 = enc_b[c*HDIM + tid];
    float w1 = enc_w[c*HDIM + tid+256], b1 = enc_b[c*HDIM + tid+256];
    float sw = w0+w1, sb = b0+b1;
    float sww = w0*w0 + w1*w1, swb = w0*b0 + w1*b1, sbb = b0*b0 + b1*b1;
    #pragma unroll
    for (int m = 1; m < 32; m <<= 1) {
        sw  += __shfl_xor(sw,  m);
        sb  += __shfl_xor(sb,  m);
        sww += __shfl_xor(sww, m);
        swb += __shfl_xor(swb, m);
        sbb += __shfl_xor(sbb, m);
    }
    int wv = tid >> 5, ln = tid & 31;
    if (ln == 0) { red[wv][0]=sw; red[wv][1]=sb; red[wv][2]=sww; red[wv][3]=swb; red[wv][4]=sbb; }
    __syncthreads();
    if (tid == 0) {
        float S[5] = {0,0,0,0,0};
        for (int i = 0; i < 8; i++)
            for (int j = 0; j < 5; j++) S[j] += red[i][j];
        float inv = 1.0f / HDIM;
        float mw = S[0]*inv, mb = S[1]*inv;
        float Vw  = S[2]*inv - mw*mw;
        float Cwb = S[3]*inv - mw*mb;
        float Vb  = S[4]*inv - mb*mb;
        fin[0]=mw; fin[1]=mb; fin[2]=Vw; fin[3]=Cwb; fin[4]=Vb;
        float* st = stats + c*8;
        st[0]=mw; st[1]=mb; st[2]=Vw; st[3]=Cwb; st[4]=Vb;
    }
    __syncthreads();
    float mw = fin[0], mb = fin[1];
    for (int j = tid; j < HDIM; j += 256) {
        int k = c*HDIM + j;
        float g = enc_gamma[k];
        gw[k]  = g * (enc_w[k] - mw);
        gb[k]  = g * (enc_b[k] - mb);
        bet[k] = enc_beta[k];
    }
}

// ---------- Prep 2: per-(token,channel) scale/shift ----------
__global__ void prep_tok_kernel(const float* __restrict__ x,
                                const float* __restrict__ stats,
                                float2* __restrict__ tok) {
    int idx = blockIdx.x * blockDim.x + threadIdx.x;   // t*8 + c
    if (idx >= TOK*CNUM) return;
    int t = idx >> 3, c = idx & 7;
    float xv = x[c*TOK + t];
    const float* st = stats + c*8;
    float var = fmaf(xv*xv, st[2], fmaf(2.0f*xv, st[3], st[4]));
    float rs = rsqrtf(var + EPSF);
    tok[idx] = make_float2(xv*rs, rs);
}

// ---------- Prep 3: mod_w -> bf16 hi/lo split ----------
__global__ void prep_w_kernel(const float* __restrict__ mw,
                              __bf16* __restrict__ hi,
                              __bf16* __restrict__ lo) {
    int idx = blockIdx.x * blockDim.x + threadIdx.x;
    if (idx >= HDIM*KDIM) return;
    float v = mw[idx];
    __bf16 h = (__bf16)v;
    hi[idx] = h;
    lo[idx] = (__bf16)(v - (float)h);
}

__device__ __forceinline__ v16bf cat16(bf16x8 a, bf16x8 b) {
    return __builtin_shufflevector(a, b, 0,1,2,3,4,5,6,7,8,9,10,11,12,13,14,15);
}

// ---------- Fused: regenerate A tiles + async-stage B -> bf16x2 WMMA -> LN -> GELU ----------
__global__ __launch_bounds__(256, 1)
void fused_gemm_kernel(const float2* __restrict__ tokscal,
                       const float* __restrict__ gw,
                       const float* __restrict__ gb,
                       const float* __restrict__ bet,
                       const __bf16* __restrict__ Bhi,
                       const __bf16* __restrict__ Blo,
                       const float* __restrict__ mod_b,
                       const float* __restrict__ mod_gamma,
                       const float* __restrict__ mod_beta,
                       float* __restrict__ out) {
    __shared__ __align__(16) __bf16 sBhi[HDIM * LDK];   // 40 KB
    __shared__ __align__(16) __bf16 sBlo[HDIM * LDK];   // 40 KB
    __shared__ __align__(16) __bf16 sAhi[MTILE * LDK];  // 5 KB
    __shared__ __align__(16) __bf16 sAlo[MTILE * LDK];  // 5 KB
    __shared__ float2 sTok[MTILE * CNUM];               // 4 KB
    __shared__ float sSum[MTILE][2];
    __shared__ float sSqs[MTILE][2];

    const int tid = threadIdx.x;
    const int t0  = blockIdx.x * MTILE;

    for (int i = tid; i < MTILE*CNUM; i += 256) sTok[i] = tokscal[t0*CNUM + i];

    const int lane  = tid & 31;
    const int wv    = tid >> 5;
    const int msub  = wv & 3;          // 16-token group 0..3
    const int nhalf = wv >> 2;         // output cols 0..255 or 256..511
    const int half  = lane >> 4;
    const int l15   = lane & 15;

    v8f acc[16];
    #pragma unroll
    for (int nt = 0; nt < 16; nt++) acc[nt] = v8f{0,0,0,0,0,0,0,0};

    // A regeneration assignment: thread -> (token, 8 consecutive k)
    const int atok = tid >> 2;         // 0..63
    const int aj0  = (tid & 3) << 3;   // 0,8,16,24
    // B staging assignment: thread copies 8 x 16B per array (512 rows x 4 segs)
    const int brow = tid >> 2;         // reused with +64 stride below
    const int bseg = (tid & 3) << 3;   // k offset 0,8,16,24
    __syncthreads();

    for (int kc = 0; kc < KDIM; kc += KCH) {
        // ---- issue async DMA of B chunk into LDS (overlaps with A VALU work) ----
        #pragma unroll
        for (int i = 0; i < 8; i++) {
            int row = brow + i*64;     // 0..511
            const __bf16* gh = Bhi + (size_t)row*KDIM + kc + bseg;
            const __bf16* gl = Blo + (size_t)row*KDIM + kc + bseg;
            cp16_async(gh, &sBhi[row*LDK + bseg]);
            cp16_async(gl, &sBlo[row*LDK + bseg]);
        }

        // ---- regenerate encoder output tile (GELU'd, bf16 hi/lo) in registers ----
        const int cch = kc >> 9;       // channel of this K chunk (32 | 512)
        float2 ab = sTok[atok*CNUM + cch];
        bf16x8 vh, vl;
        #pragma unroll
        for (int i = 0; i < 8; i++) {
            int k = kc + aj0 + i;
            float v = fmaf(ab.x, gw[k], fmaf(ab.y, gb[k], bet[k]));
            float h = gelu_exact(v);
            __bf16 hh = (__bf16)h;
            vh[i] = hh;
            vl[i] = (__bf16)(h - (float)hh);
        }
        *(bf16x8*)&sAhi[atok*LDK + aj0] = vh;
        *(bf16x8*)&sAlo[atok*LDK + aj0] = vl;

        wait_async_copies();
        __syncthreads();

        // ---- WMMA over this K chunk (all operands from LDS) ----
        {
            int abase = (msub*16 + l15)*LDK + half*8;
            v16bf aH = cat16(*(const bf16x8*)&sAhi[abase],
                             *(const bf16x8*)&sAhi[abase + 16]);
            v16bf aL = cat16(*(const bf16x8*)&sAlo[abase],
                             *(const bf16x8*)&sAlo[abase + 16]);
            int bbase = (nhalf*256 + l15)*LDK + half*8;
            #pragma unroll
            for (int nt = 0; nt < 16; nt++) {
                int boff = bbase + nt*16*LDK;
                v16bf bH = cat16(*(const bf16x8*)&sBhi[boff],
                                 *(const bf16x8*)&sBhi[boff + 16]);
                v16bf bL = cat16(*(const bf16x8*)&sBlo[boff],
                                 *(const bf16x8*)&sBlo[boff + 16]);
                acc[nt] = __builtin_amdgcn_wmma_f32_16x16x32_bf16(false, aH, false, bH, (short)0, acc[nt], false, false);
                acc[nt] = __builtin_amdgcn_wmma_f32_16x16x32_bf16(false, aH, false, bL, (short)0, acc[nt], false, false);
                acc[nt] = __builtin_amdgcn_wmma_f32_16x16x32_bf16(false, aL, false, bH, (short)0, acc[nt], false, false);
            }
        }
        __syncthreads();
    }

    // ---- epilogue: + mod_b, LayerNorm over 512 cols (cross-wave), GELU, store ----
    #pragma unroll
    for (int nt = 0; nt < 16; nt++) {
        float bias = mod_b[nhalf*256 + nt*16 + l15];
        #pragma unroll
        for (int r = 0; r < 8; r++) acc[nt][r] += bias;
    }

    #pragma unroll
    for (int r = 0; r < 8; r++) {
        float s = 0.f, q = 0.f;
        #pragma unroll
        for (int nt = 0; nt < 16; nt++) { float v = acc[nt][r]; s += v; q = fmaf(v, v, q); }
        s += __shfl_xor(s, 1);  q += __shfl_xor(q, 1);
        s += __shfl_xor(s, 2);  q += __shfl_xor(q, 2);
        s += __shfl_xor(s, 4);  q += __shfl_xor(q, 4);
        s += __shfl_xor(s, 8);  q += __shfl_xor(q, 8);
        if (l15 == 0) {
            int row = msub*16 + r + half*8;
            sSum[row][nhalf] = s;
            sSqs[row][nhalf] = q;
        }
    }
    __syncthreads();

    float mu[8], rstd[8];
    #pragma unroll
    for (int r = 0; r < 8; r++) {
        int row = msub*16 + r + half*8;
        float S = sSum[row][0] + sSum[row][1];
        float Q = sSqs[row][0] + sSqs[row][1];
        float m = S * (1.0f/HDIM);
        float var = Q * (1.0f/HDIM) - m*m;
        mu[r] = m;
        rstd[r] = rsqrtf(var + EPSF);
    }

    #pragma unroll
    for (int nt = 0; nt < 16; nt++) {
        int col = nhalf*256 + nt*16 + l15;
        float gmm = mod_gamma[col], btt = mod_beta[col];
        #pragma unroll
        for (int r = 0; r < 8; r++) {
            int row = msub*16 + r + half*8;
            float y = (acc[nt][r] - mu[r]) * rstd[r] * gmm + btt;
            out[(size_t)(t0 + row)*HDIM + col] = gelu_exact(y);
        }
    }
}

extern "C" void kernel_launch(void* const* d_in, const int* in_sizes, int n_in,
                              void* d_out, int out_size, void* d_ws, size_t ws_size,
                              hipStream_t stream) {
    const float* x         = (const float*)d_in[0];
    const float* enc_w     = (const float*)d_in[1];
    const float* enc_b     = (const float*)d_in[2];
    const float* enc_gamma = (const float*)d_in[3];
    const float* enc_beta  = (const float*)d_in[4];
    const float* mod_w     = (const float*)d_in[5];
    const float* mod_b     = (const float*)d_in[6];
    const float* mod_gamma = (const float*)d_in[7];
    const float* mod_beta  = (const float*)d_in[8];

    char* ws = (char*)d_ws;
    float*  stats = (float*)(ws + WS_STATS);
    float*  gw    = (float*)(ws + WS_GW);
    float*  gb    = (float*)(ws + WS_GB);
    float*  bet   = (float*)(ws + WS_BETA);
    float2* tok   = (float2*)(ws + WS_TOK);
    __bf16* Bhi   = (__bf16*)(ws + WS_BHI);
    __bf16* Blo   = (__bf16*)(ws + WS_BLO);

    prep_stats_kernel<<<CNUM, 256, 0, stream>>>(enc_w, enc_b, enc_gamma, enc_beta,
                                                stats, gw, gb, bet);
    prep_tok_kernel<<<(TOK*CNUM)/256, 256, 0, stream>>>(x, stats, tok);
    prep_w_kernel<<<(HDIM*KDIM)/256, 256, 0, stream>>>(mod_w, Bhi, Blo);
    fused_gemm_kernel<<<TOK/MTILE, 256, 0, stream>>>(tok, gw, gb, bet, Bhi, Blo,
                                                     mod_b, mod_gamma, mod_beta,
                                                     (float*)d_out);
}
